// GatedDeltaNetAttention_67207648248339
// MI455X (gfx1250) — matrix operand order repaired and checked by reading, not compiled, not run on hardware
//
#include <hip/hip_runtime.h>
#include <hip/hip_bf16.h>

typedef __attribute__((ext_vector_type(16))) __bf16          v16bf;
typedef __attribute__((ext_vector_type(16))) unsigned short  v16u;
typedef __attribute__((ext_vector_type(8)))  unsigned short  v8u;
typedef __attribute__((ext_vector_type(8)))  float           v8f;

#define DIMX 2048
#define TLEN 2048
#define NHEAD 16
#define DHEAD 64

__device__ __forceinline__ void wait_async0() {
    asm volatile("s_wait_asynccnt 0" ::: "memory");
}

// 16-byte async copy global -> LDS (ASYNCcnt-tracked, bypasses VGPRs).
// VDST = LDS byte address (low 32 bits of generic shared pointer),
// VADDR = 64-bit global address, GV mode (no SADDR).
__device__ __forceinline__ void cp16_g2l(const unsigned short* g, unsigned short* l) {
    unsigned int       lds_addr = (unsigned int)(uintptr_t)l;
    unsigned long long gaddr    = (unsigned long long)(uintptr_t)g;
    asm volatile("global_load_async_to_lds_b128 %0, %1, off"
                 :: "v"(lds_addr), "v"(gaddr)
                 : "memory");
}

__device__ __forceinline__ unsigned short f2bf(float f) {
    unsigned int u = __float_as_uint(f);
    u += 0x7FFFu + ((u >> 16) & 1u);   // round-to-nearest-even
    return (unsigned short)(u >> 16);
}

// ---------------- f32 -> bf16 conversion ----------------
__global__ void cvt_bf16(const float* __restrict__ in, unsigned short* __restrict__ out, int n) {
    int i = blockIdx.x * blockDim.x + threadIdx.x;
    if (i < n) out[i] = f2bf(in[i]);
}

// ---------------- bf16 WMMA GEMM: C[M,N] = A[M,K] * B[N,K]^T ----------------
// 256 threads (8 waves). Block tile 128(M) x 64(N), K step 32.
// Each wave owns a 32x32 C sub-tile: 4 accumulators, 4 WMMAs per K-step.
// Double-buffered LDS filled by async global->LDS copies (ASYNCcnt).
__global__ __launch_bounds__(256) void gemm_bf16(const unsigned short* __restrict__ A,
                                                 const unsigned short* __restrict__ B,
                                                 float* __restrict__ C,
                                                 int M, int N, int K) {
    __shared__ unsigned short As[2][128][32];
    __shared__ unsigned short Bs[2][64][32];
    const int tid  = threadIdx.x;
    const int lane = tid & 31;
    const int wave = tid >> 5;
    const int wr   = wave >> 1;        // 0..3 -> 32-row slice
    const int wc   = wave & 1;         // 0..1 -> 32-col slice
    const int m0   = blockIdx.y * 128;
    const int n0   = blockIdx.x * 64;
    const int Ms   = wr * 32;
    const int Ns   = wc * 32;
    const int mrow = lane & 15;
    const int half = lane >> 4;

    v8f acc00 = {}, acc01 = {}, acc10 = {}, acc11 = {};

    // stage one 128x32 A tile + 64x32 B tile into LDS buffer b
    auto load_tiles = [&](int b, int kt) {
#pragma unroll
        for (int j = 0; j < 2; ++j) {                 // A: 512 x 16B chunks
            int e  = tid + j * 256;
            int r  = e >> 2;
            int c8 = (e & 3) << 3;
            cp16_g2l(&A[(size_t)(m0 + r) * K + kt + c8], &As[b][r][c8]);
        }
        {                                             // B: 256 x 16B chunks
            int r  = tid >> 2;
            int c8 = (tid & 3) << 3;
            cp16_g2l(&B[(size_t)(n0 + r) * K + kt + c8], &Bs[b][r][c8]);
        }
    };

    load_tiles(0, 0);
    wait_async0();
    __syncthreads();

    int buf = 0;
    for (int kt = 0; kt < K; kt += 32) {
        if (kt + 32 < K) load_tiles(buf ^ 1, kt + 32);   // overlap next tile fill

        // A fragments: rows Ms..Ms+15 and Ms+16..Ms+31
        // lane(m=lane%16, half): elems 0..7 -> K=half*8+i, 8..15 -> K=16+half*8+(i-8)
        const unsigned short* pa0 = &As[buf][Ms + mrow][half * 8];
        const unsigned short* pa1 = &As[buf][Ms + 16 + mrow][half * 8];
        v8u a0lo = *(const v8u*)pa0,  a0hi = *(const v8u*)(pa0 + 16);
        v8u a1lo = *(const v8u*)pa1,  a1hi = *(const v8u*)(pa1 + 16);
        v16bf a0 = __builtin_bit_cast(v16bf,
            __builtin_shufflevector(a0lo, a0hi, 0,1,2,3,4,5,6,7,8,9,10,11,12,13,14,15));
        v16bf a1 = __builtin_bit_cast(v16bf,
            __builtin_shufflevector(a1lo, a1hi, 0,1,2,3,4,5,6,7,8,9,10,11,12,13,14,15));

        // B fragments: cols Ns..Ns+15 and Ns+16..Ns+31; elems 0..7 -> K=half*8+i, 8..15 -> K=16+half*8+(i-8)
        const unsigned short* pb0 = &Bs[buf][Ns + mrow][half * 8];
        const unsigned short* pb1 = &Bs[buf][Ns + 16 + mrow][half * 8];
        v16bf b0 = __builtin_bit_cast(v16bf, __builtin_shufflevector(*(const v8u*)pb0, *(const v8u*)(pb0 + 16), 0,1,2,3,4,5,6,7,8,9,10,11,12,13,14,15));
        v16bf b1 = __builtin_bit_cast(v16bf, __builtin_shufflevector(*(const v8u*)pb1, *(const v8u*)(pb1 + 16), 0,1,2,3,4,5,6,7,8,9,10,11,12,13,14,15));

        acc00 = __builtin_amdgcn_wmma_f32_16x16x32_bf16(false, a0, false, b0, (short)0, acc00, false, false);
        acc01 = __builtin_amdgcn_wmma_f32_16x16x32_bf16(false, a0, false, b1, (short)0, acc01, false, false);
        acc10 = __builtin_amdgcn_wmma_f32_16x16x32_bf16(false, a1, false, b0, (short)0, acc10, false, false);
        acc11 = __builtin_amdgcn_wmma_f32_16x16x32_bf16(false, a1, false, b1, (short)0, acc11, false, false);

        wait_async0();                     // next tile landed in LDS
        __syncthreads();
        buf ^= 1;
    }

    const int cn = lane & 15;
#pragma unroll
    for (int r = 0; r < 8; ++r) {
        int mmA = m0 + Ms + r + half * 8;          // rows of a0-based accs
        int mmB = m0 + Ms + 16 + r + half * 8;     // rows of a1-based accs
        C[(size_t)mmA * N + n0 + Ns + cn]      = acc00[r];
        C[(size_t)mmA * N + n0 + Ns + 16 + cn] = acc01[r];
        C[(size_t)mmB * N + n0 + Ns + cn]      = acc10[r];
        C[(size_t)mmB * N + n0 + Ns + 16 + cn] = acc11[r];
    }
}

// ---------------- causal depthwise conv (K=4) + SiLU ----------------
__global__ void conv_silu(const float* __restrict__ pre, const float* __restrict__ w,
                          float* __restrict__ out) {
    int idx = blockIdx.x * blockDim.x + threadIdx.x;   // T*CONV_DIM = 2048*2048
    if (idx >= TLEN * DIMX) return;
    int t = idx >> 11, c = idx & (DIMX - 1);
    float acc = 0.f;
#pragma unroll
    for (int j = 0; j < 4; ++j) {
        int ts = t - 3 + j;
        float xv = (ts >= 0) ? pre[(size_t)ts * DIMX + c] : 0.f;
        acc += w[c * 4 + j] * xv;
    }
    out[idx] = acc / (1.f + __expf(-acc));
}

// ---------------- beta = sigmoid(xWb), g = -exp(A_log)*softplus(xWa + dt_bias) ----------------
__global__ __launch_bounds__(256) void proj_bg(const float* __restrict__ x,
                                               const float* __restrict__ Wb,
                                               const float* __restrict__ Wa,
                                               const float* __restrict__ dt_bias,
                                               const float* __restrict__ A_log,
                                               float* __restrict__ beta,
                                               float* __restrict__ g) {
    __shared__ float xl[DIMX];
    __shared__ float red[32][8];
    int t = blockIdx.x, tid = threadIdx.x;
#pragma unroll
    for (int j = 0; j < 8; ++j) xl[tid + j * 256] = x[(size_t)t * DIMX + tid + j * 256];
    __syncthreads();
    int grp = tid >> 3, sub = tid & 7;   // 32 outputs x 8 partials
    const float* Wrow = (grp < 16) ? &Wb[grp * DIMX] : &Wa[(grp - 16) * DIMX];
    float acc = 0.f;
    for (int i = sub; i < DIMX; i += 8) acc += xl[i] * Wrow[i];
    red[grp][sub] = acc;
    __syncthreads();
    if (sub == 0) {
        float s = 0.f;
#pragma unroll
        for (int j = 0; j < 8; ++j) s += red[grp][j];
        if (grp < 16) {
            beta[t * NHEAD + grp] = 1.f / (1.f + __expf(-s));
        } else {
            int h = grp - 16;
            float aa = s + dt_bias[h];
            float sp = (aa > 20.f) ? aa : log1pf(__expf(aa));
            g[t * NHEAD + h] = -__expf(A_log[h]) * sp;
        }
    }
}

// ---------------- q/k L2-norm with GQA head replication ----------------
__global__ __launch_bounds__(256) void norm_qk(const float* __restrict__ qkv,
                                               float* __restrict__ qn,
                                               float* __restrict__ kn) {
    int tid = threadIdx.x;
    int w = blockIdx.x * 8 + (tid >> 5);   // one warp per (t,h); 2048*16 warps
    int lane = tid & 31;
    int t = w >> 4, h = w & 15, hs = h >> 1;
    {
        const float* src = &qkv[(size_t)t * DIMX + hs * 64];
        float v0 = src[lane], v1 = src[lane + 32];
        float s = v0 * v0 + v1 * v1;
        for (int off = 16; off; off >>= 1) s += __shfl_xor(s, off, 32);
        float r = rsqrtf(s + 1e-6f) * 0.125f;   // * DK^-0.5
        float* dst = &qn[(size_t)(t * NHEAD + h) * 64];
        dst[lane] = v0 * r; dst[lane + 32] = v1 * r;
    }
    {
        const float* src = &qkv[(size_t)t * DIMX + 512 + hs * 64];
        float v0 = src[lane], v1 = src[lane + 32];
        float s = v0 * v0 + v1 * v1;
        for (int off = 16; off; off >>= 1) s += __shfl_xor(s, off, 32);
        float r = rsqrtf(s + 1e-6f);
        float* dst = &kn[(size_t)(t * NHEAD + h) * 64];
        dst[lane] = v0 * r; dst[lane + 32] = v1 * r;
    }
}

// ---------------- sequential gated delta rule: one block per head ----------------
__global__ __launch_bounds__(256) void delta_rule(const float* __restrict__ qn,
                                                  const float* __restrict__ kn,
                                                  const float* __restrict__ qkv,   // v at col 1024
                                                  const float* __restrict__ g,
                                                  const float* __restrict__ beta,
                                                  float* __restrict__ y) {
    const int h = blockIdx.x;
    const int tid = threadIdx.x;
    const int vcol = tid & 63;          // dv index
    const int kq = tid >> 6;            // 0..3
    const int k0 = kq * 16;             // this thread's dk range
    float S[16];
#pragma unroll
    for (int i = 0; i < 16; ++i) S[i] = 0.f;

    __shared__ float sk[64], sq[64], sv[64];
    __shared__ float red[256];

    for (int t = 0; t < TLEN; ++t) {
        const int th = t * NHEAD + h;
        if (tid < 64)       sk[tid]        = kn[(size_t)th * 64 + tid];
        else if (tid < 128) sq[tid - 64]   = qn[(size_t)th * 64 + (tid - 64)];
        else if (tid < 192) sv[tid - 128]  = qkv[(size_t)t * DIMX + 1024 + h * 64 + (tid - 128)];
        float eg = __expf(g[th]);
        float bt = beta[th];
        __syncthreads();

        float p = 0.f;
#pragma unroll
        for (int i = 0; i < 16; ++i) p += sk[k0 + i] * S[i];
        red[tid] = p;
        __syncthreads();

        float pred  = eg * (red[vcol] + red[vcol + 64] + red[vcol + 128] + red[vcol + 192]);
        float delta = (sv[vcol] - pred) * bt;
        float o = 0.f;
#pragma unroll
        for (int i = 0; i < 16; ++i) {
            S[i] = S[i] * eg + sk[k0 + i] * delta;
            o += sq[k0 + i] * S[i];
        }
        red[tid] = o;
        __syncthreads();

        if (tid < 64)
            y[(size_t)th * 64 + tid] = red[tid] + red[tid + 64] + red[tid + 128] + red[tid + 192];
    }
}

// ---------------- gated RMSNorm + silu(z) gate, emit bf16 for final GEMM ----------------
__global__ __launch_bounds__(256) void gated_norm(const float* __restrict__ y,
                                                  const float* __restrict__ z,
                                                  const float* __restrict__ nw,
                                                  unsigned short* __restrict__ yn_bf) {
    int tid = threadIdx.x;
    int w = blockIdx.x * 8 + (tid >> 5);
    int lane = tid & 31;
    int t = w >> 4, h = w & 15;
    const float* yr = &y[(size_t)(t * NHEAD + h) * 64];
    float y0 = yr[lane], y1 = yr[lane + 32];
    float s = y0 * y0 + y1 * y1;
    for (int off = 16; off; off >>= 1) s += __shfl_xor(s, off, 32);
    float r = rsqrtf(s * (1.f / 64.f) + 1e-6f);
    const float* zr = &z[(size_t)t * 1024 + h * 64];
    float z0 = zr[lane], z1 = zr[lane + 32];
    float g0 = z0 / (1.f + __expf(-z0));
    float g1 = z1 / (1.f + __expf(-z1));
    unsigned short* o = &yn_bf[(size_t)t * 1024 + h * 64];
    o[lane]      = f2bf(nw[lane]      * y0 * r * g0);
    o[lane + 32] = f2bf(nw[lane + 32] * y1 * r * g1);
}

extern "C" void kernel_launch(void* const* d_in, const int* in_sizes, int n_in,
                              void* d_out, int out_size, void* d_ws, size_t ws_size,
                              hipStream_t stream) {
    const float* x      = (const float*)d_in[0];   // [T, 2048]
    const float* W_qkv  = (const float*)d_in[1];   // [2048, 2048]
    const float* W_z    = (const float*)d_in[2];   // [1024, 2048]
    const float* W_b    = (const float*)d_in[3];   // [16, 2048]
    const float* W_a    = (const float*)d_in[4];   // [16, 2048]
    const float* W_out  = (const float*)d_in[5];   // [2048, 1024]
    const float* conv_w = (const float*)d_in[6];   // [2048, 1, 4]
    const float* dtb    = (const float*)d_in[7];   // [16]
    const float* A_log  = (const float*)d_in[8];   // [16]
    const float* nw     = (const float*)d_in[9];   // [64]
    float* out = (float*)d_out;                    // [T, 2048]

    char* ws = (char*)d_ws;
    size_t off = 0;
    auto carve = [&](size_t bytes) { char* p = ws + off; off += (bytes + 255) & ~(size_t)255; return p; };
    unsigned short* x_bf    = (unsigned short*)carve((size_t)TLEN * DIMX * 2);
    unsigned short* Wqkv_bf = (unsigned short*)carve((size_t)DIMX * DIMX * 2);
    unsigned short* Wz_bf   = (unsigned short*)carve((size_t)1024 * DIMX * 2);
    unsigned short* Wout_bf = (unsigned short*)carve((size_t)DIMX * 1024 * 2);
    float* qkv_pre = (float*)carve((size_t)TLEN * DIMX * 4);
    float* qkv_act = (float*)carve((size_t)TLEN * DIMX * 4);
    float* zbuf    = (float*)carve((size_t)TLEN * 1024 * 4);
    float* beta    = (float*)carve((size_t)TLEN * NHEAD * 4);
    float* gbuf    = (float*)carve((size_t)TLEN * NHEAD * 4);
    float* qn      = (float*)carve((size_t)TLEN * NHEAD * 64 * 4);
    float* kn      = (float*)carve((size_t)TLEN * NHEAD * 64 * 4);
    float* ybuf    = (float*)carve((size_t)TLEN * NHEAD * 64 * 4);
    unsigned short* yn_bf = (unsigned short*)carve((size_t)TLEN * 1024 * 2);

    // 1) bf16 conversions
    cvt_bf16<<<(TLEN * DIMX + 255) / 256, 256, 0, stream>>>(x, x_bf, TLEN * DIMX);
    cvt_bf16<<<(DIMX * DIMX + 255) / 256, 256, 0, stream>>>(W_qkv, Wqkv_bf, DIMX * DIMX);
    cvt_bf16<<<(1024 * DIMX + 255) / 256, 256, 0, stream>>>(W_z, Wz_bf, 1024 * DIMX);
    cvt_bf16<<<(DIMX * 1024 + 255) / 256, 256, 0, stream>>>(W_out, Wout_bf, DIMX * 1024);

    // 2) qkv projection: [2048,2048] = x @ W_qkv^T
    gemm_bf16<<<dim3(DIMX / 64, TLEN / 128), 256, 0, stream>>>(x_bf, Wqkv_bf, qkv_pre, TLEN, DIMX, DIMX);
    // 3) z projection: [2048,1024]
    gemm_bf16<<<dim3(1024 / 64, TLEN / 128), 256, 0, stream>>>(x_bf, Wz_bf, zbuf, TLEN, 1024, DIMX);
    // 4) beta / g scalar projections (f32)
    proj_bg<<<TLEN, 256, 0, stream>>>(x, W_b, W_a, dtb, A_log, beta, gbuf);
    // 5) causal depthwise conv + silu
    conv_silu<<<(TLEN * DIMX + 255) / 256, 256, 0, stream>>>(qkv_pre, conv_w, qkv_act);
    // 6) q/k l2norm (+GQA expand)
    norm_qk<<<TLEN * NHEAD / 8, 256, 0, stream>>>(qkv_act, qn, kn);
    // 7) sequential gated delta rule
    delta_rule<<<NHEAD, 256, 0, stream>>>(qn, kn, qkv_act, gbuf, beta, ybuf);
    // 8) gated RMSNorm -> bf16
    gated_norm<<<TLEN * NHEAD / 8, 256, 0, stream>>>(ybuf, zbuf, nw, yn_bf);
    // 9) output projection: out = yn @ W_out^T  [2048,2048], K=1024
    gemm_bf16<<<dim3(DIMX / 64, TLEN / 128), 256, 0, stream>>>(yn_bf, Wout_bf, out, TLEN, DIMX, 1024);
}
